// StyleConv2d_8108898255007
// MI455X (gfx1250) — matrix operand order repaired and checked by reading, not compiled, not run on hardware
//
#include <hip/hip_runtime.h>

typedef __attribute__((ext_vector_type(16))) _Float16 v16h;
typedef __attribute__((ext_vector_type(8)))  _Float16 v8h;
typedef __attribute__((ext_vector_type(8)))  float    v8f;
typedef __attribute__((ext_vector_type(4)))  float    v4f;

#define B_    16
#define CIN   256
#define COUT  256
#define H_    64
#define W_    64
#define LAT   512
#define KK    9              // 3x3 taps
#define KTOT  (KK * CIN)     // 2304 = GEMM K
#define XT_Y  66             // padded H
#define XT_X  72             // padded W (rounded for alignment)

// workspace layout (bytes)
#define S_OFF    0                          // s: 16*256 f32            = 16 KB
#define WMOD_OFF 16384                      // wmod f16 [b][cout][2304] = 18.87 MB
#define XT_OFF   (16384 + 18874368)         // xt f16 [b][66][72][256]  = 38.93 MB
// total ~55.1 MB

// ---------------- 1) style affine: s = style @ style_w + style_b ----------------
__global__ void k_style(const float* __restrict__ style, const float* __restrict__ style_w,
                        const float* __restrict__ style_b, float* __restrict__ s_out) {
    const int b = blockIdx.x;        // 16
    const int c = threadIdx.x;       // 256
    const float* st = style + b * LAT;
    float acc = style_b[c];
#pragma unroll 8
    for (int l = 0; l < LAT; ++l) acc += st[l] * style_w[l * CIN + c];
    s_out[b * CIN + c] = acc;
}

// ---------------- 2a) zero the padded channels-last x tensor ----------------
__global__ void k_zero(v8h* __restrict__ p, long n8) {
    long i = (long)blockIdx.x * 256 + threadIdx.x;
    if (i < n8) { v8h z = {}; p[i] = z; }
}

// ---------------- 2b) pack: x f32 NCHW -> xt f16 [b][y+1][x+1][c] ----------------
__global__ void k_pack(const float* __restrict__ x, _Float16* __restrict__ xt) {
    const int y = blockIdx.x;        // 0..63
    const int b = blockIdx.y;        // 0..15
    const int tid = threadIdx.x;     // 256
    __shared__ _Float16 tile[64 * 32];   // [x][c] for one 32-channel slab

    for (int cblk = 0; cblk < 8; ++cblk) {
        const int cl = tid >> 3;     // 0..31 channel in slab
        const int xq = tid & 7;      // 0..7  8-col segment
        const float* src = x + (((long)(b * CIN + cblk * 32 + cl) * H_ + y) * W_ + xq * 8);
        v4f f0 = *(const v4f*)src;
        v4f f1 = *(const v4f*)(src + 4);
        const int xs = xq * 8;
        tile[(xs + 0) * 32 + cl] = (_Float16)f0.x;
        tile[(xs + 1) * 32 + cl] = (_Float16)f0.y;
        tile[(xs + 2) * 32 + cl] = (_Float16)f0.z;
        tile[(xs + 3) * 32 + cl] = (_Float16)f0.w;
        tile[(xs + 4) * 32 + cl] = (_Float16)f1.x;
        tile[(xs + 5) * 32 + cl] = (_Float16)f1.y;
        tile[(xs + 6) * 32 + cl] = (_Float16)f1.z;
        tile[(xs + 7) * 32 + cl] = (_Float16)f1.w;
        __syncthreads();
        const int row = tid >> 2;    // 0..63 column of image
        const int q   = tid & 3;     // 8-half segment of slab
        v8h v = *(const v8h*)&tile[row * 32 + q * 8];
        _Float16* dst = xt + ((((long)b * XT_Y + (y + 1)) * XT_X + (row + 1)) * CIN
                              + cblk * 32 + q * 8);
        *(v8h*)dst = v;
        __syncthreads();
    }
}

// ---------------- 3) modulate + demodulate -> f16 A-matrix [b][cout][r*256+cin] ----------------
__global__ void k_mod(const float* __restrict__ w, const float* __restrict__ s,
                      _Float16* __restrict__ wmod) {
    const int cout = blockIdx.x;     // 256
    const int b    = blockIdx.y;     // 16
    const int cin  = threadIdx.x;    // 256
    const float sv = s[b * CIN + cin];
    const float* wr = w + ((long)cout * CIN + cin) * KK;   // [cout][cin][3][3]
    float wl[KK];
    float acc = 0.f;
#pragma unroll
    for (int j = 0; j < KK; ++j) { float t = wr[j] * sv; wl[j] = t; acc += t * t; }
    __shared__ float red[256];
    red[cin] = acc;
    __syncthreads();
    for (int st = 128; st > 0; st >>= 1) {
        if (cin < st) red[cin] += red[cin + st];
        __syncthreads();
    }
    const float d = __frsqrt_rn(red[0] + 1e-8f);
    _Float16* o = wmod + (long)(b * COUT + cout) * KTOT + cin;
#pragma unroll
    for (int j = 0; j < KK; ++j) o[j * CIN] = (_Float16)(wl[j] * d);   // K = r*256+cin
}

// ---------------- 4) implicit-GEMM conv via v_wmma_f32_16x16x32_f16 ----------------
// WG = 256 threads (8 wave32). Output tile: 128 cout x 128 pixels (2 image rows).
// wave(wm 0..1, wn 0..3): 64 cout x 32 pixels = 4x2 WMMA tiles.
// Staging uses CDNA5 async global->LDS DMA (ASYNCcnt) with double-buffered
// A-tiles (per K-step lookahead) and X-chunks (per cin-block lookahead).
#define ALD 40     // padded LDS row stride (halves): 80B, 16B-aligned, conflict-spread
#define XCOLS 66   // only columns actually consumed by 3x3 taps
#define XUNITS (4 * XCOLS * 4)   // 1056 b128 units per X chunk

__global__ void __launch_bounds__(256)
k_conv(const _Float16* __restrict__ wmod, const _Float16* __restrict__ xt,
       const float* __restrict__ noise, const float* __restrict__ bias,
       const float* __restrict__ nw_p, float* __restrict__ out) {
    const int pt  = blockIdx.x;      // 0..31 pixel tile (rows 2pt..2pt+1)
    const int ct  = blockIdx.y;      // 0..1  cout tile (128)
    const int b   = blockIdx.z;      // 0..15
    const int tid = threadIdx.x;
    const int lane = tid & 31;
    const int wave = tid >> 5;
    const int wm = wave & 1;
    const int wn = wave >> 1;
    const int lm = lane & 15;
    const int lh = lane >> 4;        // half-wave select per ISA fragment layout

    __shared__ _Float16 Xl[2][4 * XCOLS * ALD];  // [buf][dy][x 0..65][c] = 2 x 21120 B
    __shared__ _Float16 Al[2][128 * ALD];        // [buf][cout][c]        = 2 x 10240 B

    const int  coutBase = ct * 128;
    const long wbase = (long)(b * COUT + coutBase) * KTOT;
    const long xbase = ((long)b * XT_Y + 2 * pt) * XT_X * CIN;

    // issue async DMA: one 32-K A tile (128 cout x 32) for GEMM step 'step'
    auto issueA = [&](int step, int buf) {
        const int cb = step / KK, r = step - cb * KK;
#pragma unroll
        for (int i = 0; i < 2; ++i) {
            int u  = tid * 2 + i;                // 512 b128 units
            int co = u >> 2, q = u & 3;
            const _Float16* src = wmod + wbase + (long)co * KTOT + r * CIN + cb * 32 + q * 8;
            unsigned dst = (unsigned)(unsigned long long)&Al[buf][co * ALD + q * 8];
            asm volatile("global_load_async_to_lds_b128 %0, %1, off"
                         :: "v"(dst), "v"(src) : "memory");
        }
    };
    // issue async DMA: X chunk (4 rows x 66 cols x 32 ch) for cin-block 'cb'
    auto issueX = [&](int cb, int buf) {
#pragma unroll
        for (int i = 0; i < 5; ++i) {
            int u = tid + i * 256;
            if (u < XUNITS) {
                int dy  = u / (XCOLS * 4);
                int rem = u - dy * (XCOLS * 4);
                int xx  = rem >> 2, q = rem & 3;
                const _Float16* src = xt + xbase + ((long)dy * XT_X + xx) * CIN + cb * 32 + q * 8;
                unsigned dst = (unsigned)(unsigned long long)&Xl[buf][(dy * XCOLS + xx) * ALD + q * 8];
                asm volatile("global_load_async_to_lds_b128 %0, %1, off"
                             :: "v"(dst), "v"(src) : "memory");
            }
        }
    };

    v8f acc[4][2];
#pragma unroll
    for (int i = 0; i < 4; ++i)
#pragma unroll
        for (int j = 0; j < 2; ++j) { v8f z = {}; acc[i][j] = z; }

    // prologue: stage step 0 (A tile + first X chunk)
    issueX(0, 0);
    issueA(0, 0);

    for (int s = 0; s < 8 * KK; ++s) {           // 72 K-steps of 32
        asm volatile("s_wait_asynccnt 0x0" ::: "memory");   // this wave's DMA landed
        __syncthreads();                          // everyone's DMA landed, prev compute done

        // lookahead-1: stage next step while computing this one
        if (s + 1 < 8 * KK) {
            issueA(s + 1, (s + 1) & 1);
            if (((s + 1) % KK) == 0) {            // crossing into next cin-block
                int ncb = (s + 1) / KK;
                issueX(ncb, ncb & 1);
            }
        }

        const int cb = s / KK, r = s - cb * KK;
        const int ky = r / 3, kx = r - 3 * ky;
        const _Float16* Ab = Al[s & 1];
        const _Float16* Xb = Xl[cb & 1];

        // A fragments: lanes 0-15 K{0-7,16-23}, lanes 16-31 K{8-15,24-31}
        v16h af[4];
#pragma unroll
        for (int tm = 0; tm < 4; ++tm) {
            int row = wm * 64 + tm * 16 + lm;
            int k0  = lh * 8;
            v8h lo = *(const v8h*)&Ab[row * ALD + k0];
            v8h hi = *(const v8h*)&Ab[row * ALD + k0 + 16];
            af[tm] = __builtin_shufflevector(lo, hi, 0,1,2,3,4,5,6,7,8,9,10,11,12,13,14,15);
        }
        // B fragments: lanes 0-15 K{0-7,16-23}, lanes 16-31 K{8-15,24-31}; N = lane&15
        v16h bf[2];
#pragma unroll
        for (int tn = 0; tn < 2; ++tn) {
            int p  = wn * 32 + tn * 16 + lm;
            int dy = (p >> 6) + ky;
            int xc = (p & 63) + kx;
            int k0 = lh * 8;
            v8h lo = *(const v8h*)&Xb[(dy * XCOLS + xc) * ALD + k0];
            v8h hi = *(const v8h*)&Xb[(dy * XCOLS + xc) * ALD + k0 + 16];
            bf[tn] = __builtin_shufflevector(lo, hi, 0,1,2,3,4,5,6,7,8,9,10,11,12,13,14,15);
        }
#pragma unroll
        for (int tm = 0; tm < 4; ++tm)
#pragma unroll
            for (int tn = 0; tn < 2; ++tn)
                acc[tm][tn] = __builtin_amdgcn_wmma_f32_16x16x32_f16(
                    false, af[tm], false, bf[tn], (short)0, acc[tm][tn], false, false);
    }

    // ---- epilogue: noise + bias + leaky_relu * sqrt(2), straight from C/D layout ----
    const float nw = nw_p[0];
#pragma unroll
    for (int tm = 0; tm < 4; ++tm) {
#pragma unroll
        for (int tn = 0; tn < 2; ++tn) {
            int p  = wn * 32 + tn * 16 + lm;
            int yo = 2 * pt + (p >> 6);
            int xo = p & 63;
            float nz = noise[((long)b * H_ + yo) * W_ + xo] * nw;
            int cm0 = coutBase + wm * 64 + tm * 16 + lh * 8;   // VGPR i -> M = i (+8 hi half)
#pragma unroll
            for (int i = 0; i < 8; ++i) {
                int cm = cm0 + i;
                float v = acc[tm][tn][i] + nz + bias[cm];
                v = (v > 0.f ? v : 0.2f * v) * 1.41421356f;
                out[(((long)b * COUT + cm) * H_ + yo) * W_ + xo] = v;
            }
        }
    }
}

// ---------------- host launch ----------------
extern "C" void kernel_launch(void* const* d_in, const int* in_sizes, int n_in,
                              void* d_out, int out_size, void* d_ws, size_t ws_size,
                              hipStream_t stream) {
    (void)in_sizes; (void)n_in; (void)out_size; (void)ws_size;
    const float* x        = (const float*)d_in[0];
    const float* style    = (const float*)d_in[1];
    const float* noise    = (const float*)d_in[2];
    const float* weight   = (const float*)d_in[3];
    const float* style_w  = (const float*)d_in[4];
    const float* style_b  = (const float*)d_in[5];
    const float* bias     = (const float*)d_in[6];
    const float* nw       = (const float*)d_in[7];
    float* out = (float*)d_out;

    char* ws = (char*)d_ws;
    float*     s    = (float*)(ws + S_OFF);
    _Float16*  wmod = (_Float16*)(ws + WMOD_OFF);
    _Float16*  xt   = (_Float16*)(ws + XT_OFF);

    k_style<<<dim3(B_), dim3(CIN), 0, stream>>>(style, style_w, style_b, s);

    long n8 = (long)B_ * XT_Y * XT_X * CIN / 8;
    k_zero<<<dim3((unsigned)((n8 + 255) / 256)), dim3(256), 0, stream>>>((v8h*)xt, n8);
    k_pack<<<dim3(H_, B_), dim3(256), 0, stream>>>(x, xt);
    k_mod<<<dim3(COUT, B_), dim3(256), 0, stream>>>(weight, s, wmod);

    k_conv<<<dim3(32, 2, B_), dim3(256), 0, stream>>>(wmod, xt, noise, bias, nw, out);
}
